// TranAD_Basic_2439541424429
// MI455X (gfx1250) — matrix operand order repaired and checked by reading, not compiled, not run on hardware
//
#include <hip/hip_runtime.h>

typedef __attribute__((ext_vector_type(16))) _Float16 v16h;
typedef __attribute__((ext_vector_type(8)))  float    v8f;

#define FEATS  38
#define NHEAD  38
#define WINDOW 10
#define DFF    16
#define BATCHN 16384
#define BT     8            // batch elements per block (one wave)
#define ROWS   80           // WINDOW * BT  (GEMM M dimension)
#define MTILES 5            // ROWS / 16
#define LDA    40           // f32 activation row stride (dwords)
#define LDQ    120          // qkv row stride (dwords)
#define LDH    64           // f16 GEMM-A row stride (halves), K padded to 64
#define WROWS  128          // max padded N across all GEMMs (114 -> 128)
#define NTHR   32

#define UNIF(x) x = __builtin_amdgcn_readfirstlane(x)

struct Params {
  const float *src, *tgt;
  const float *enc_in_w, *enc_in_b, *enc_out_w, *enc_out_b;
  const float *enc_lin1_w, *enc_lin1_b, *enc_lin2_w, *enc_lin2_b;
  const float *enc_ln1_w, *enc_ln1_b, *enc_ln2_w, *enc_ln2_b;
  const float *dec_sa_in_w, *dec_sa_in_b, *dec_sa_out_w, *dec_sa_out_b;
  const float *dec_ca_in_w, *dec_ca_in_b, *dec_ca_out_w, *dec_ca_out_b;
  const float *dec_lin1_w, *dec_lin1_b, *dec_lin2_w, *dec_lin2_b;
  const float *dec_ln1_w, *dec_ln1_b, *dec_ln2_w, *dec_ln2_b, *dec_ln3_w, *dec_ln3_b;
  float *out;
};

// ---- stage src/tgt tile into LDS (f32), optional sqrt(D)*x + pos-encoding ----
__device__ __attribute__((noinline))
void load_tile(const float* __restrict__ g, float* As, int bbase, int with_pos)
{
  UNIF(bbase); UNIF(with_pos);
  for (int idx = (int)threadIdx.x; idx < ROWS * FEATS; idx += NTHR) {
    const int r = idx / FEATS, c = idx - r * FEATS;
    const int s = r / BT, b = r - s * BT;
    float v = g[(s * BATCHN + bbase + b) * FEATS + c];
    if (with_pos) {
      const float dv = __expf((float)c * -0.2423773782098996f); // -ln(1e4)/38
      float sn, cs;
      __sincosf((float)s * dv, &sn, &cs);
      v = v * 6.164414002968976f + sn + cs;                     // sqrt(38)
    }
    As[r * LDA + c] = v;
  }
}

// ---- f32 LDS -> f16 LDS staging of GEMM A operand, 2 halves packed per dword ----
__device__ __attribute__((noinline))
void cast_rows(const float* src, int Kact, _Float16* dst)
{
  UNIF(Kact);
  unsigned* d32 = (unsigned*)dst;
  for (int d = (int)threadIdx.x; d < ROWS * (LDH / 2); d += NTHR) {
    const int r = d >> 5;
    const int c = (d & 31) << 1;
    const int c0 = (c < Kact) ? c : 0;          // clamp address (LDS-safe), select value
    const int c1 = (c + 1 < Kact) ? (c + 1) : 0;
    const float v0 = src[r * LDA + c0];
    const float v1 = src[r * LDA + c1];
    union { _Float16 h[2]; unsigned u; } pk;
    pk.h[0] = (_Float16)((c < Kact) ? v0 : 0.f);
    pk.h[1] = (_Float16)((c + 1 < Kact) ? v1 : 0.f);
    d32[d] = pk.u;
  }
}

// ---- stage weight matrix (N,K) row-major f32 global -> f16 LDS in B-fragment
//      layout sW[n][k] (K zero-padded to 64, N zero-padded to nrows) and bias ->
//      f32 LDS (zero-padded). Branchless clamped addresses + value selects. ----
__device__ __attribute__((noinline))
void stage_w(const float* __restrict__ W, const float* __restrict__ bias,
             int Kact, int Nact, int nrows, _Float16* sW, float* sB)
{
  UNIF(Kact); UNIF(Nact); UNIF(nrows);
  unsigned* dst = (unsigned*)sW;
  const int total = nrows * 32;                 // dwords
  for (int base = (int)threadIdx.x * 8; base < total; base += NTHR * 8) {
    float v[16];
    #pragma unroll
    for (int t = 0; t < 8; ++t) {
      const int d = base + t;
      const int n = d >> 5;
      const int k = (d & 31) << 1;
      const int nc = (n < Nact) ? n : (Nact - 1);
      const int k0 = (k < Kact) ? k : (Kact - 1);
      const int k1 = (k + 1 < Kact) ? (k + 1) : (Kact - 1);
      v[2 * t]     = W[nc * Kact + k0];
      v[2 * t + 1] = W[nc * Kact + k1];
    }
    #pragma unroll
    for (int t = 0; t < 8; ++t) {
      const int d = base + t;
      const int n = d >> 5;
      const int k = (d & 31) << 1;
      union { _Float16 h[2]; unsigned u; } pk;
      pk.h[0] = (_Float16)((n < Nact && k < Kact) ? v[2 * t] : 0.f);
      pk.h[1] = (_Float16)((n < Nact && k + 1 < Kact) ? v[2 * t + 1] : 0.f);
      dst[d] = pk.u;
    }
  }
  for (int n = (int)threadIdx.x; n < nrows; n += NTHR) {
    const float bv = bias[(n < Nact) ? n : (Nact - 1)];
    sB[n] = (n < Nact) ? bv : 0.f;
  }
}

// ---- 80xN = (80xK f16) @ W^T + bias, f32 accumulate, v_wmma_f32_16x16x32_f16.
//      All operands from LDS; all 5 A-fragments preloaded so DS latency overlaps
//      the WMMA chain (partial dscnt waits instead of full drains). ----
__device__ __attribute__((noinline))
void gemm80h(const _Float16* Ah, const _Float16* Wh, const float* sB,
             int Kact, int Nact, float* Os, int ldo, int relu)
{
  UNIF(Kact); UNIF(Nact); UNIF(ldo); UNIF(relu);
  const int lane = (int)threadIdx.x;
  const int hf = lane >> 4, lr = lane & 15;
  const int ksteps = (Kact + 31) >> 5;
  const int ntiles = (Nact + 15) >> 4;
  const unsigned* Au = (const unsigned*)Ah;
  const unsigned* Wu = (const unsigned*)Wh;

  for (int nt = 0; nt < ntiles; ++nt) {
    const int n = nt * 16 + lr;
    v8f zacc = {};
    v8f acc[MTILES] = {zacc, zacc, zacc, zacc, zacc};

    for (int ks = 0; ks < ksteps; ++ks) {
      // B fragment (32x16): halves j -> K = 32*ks + 16*hf + j, column n
      const unsigned* Wr = Wu + n * (LDH / 2) + (ks << 4) + (hf << 3);
      union { v16h v; unsigned u[8]; } Bf;
      #pragma unroll
      for (int vv = 0; vv < 8; ++vv) Bf.u[vv] = Wr[vv];

      // A fragments (16x32) for all 5 M-tiles: halves j -> K = 32*ks + 16*hf + j
      union { v16h v; unsigned u[8]; } Af[MTILES];
      #pragma unroll
      for (int m = 0; m < MTILES; ++m) {
        const unsigned* Ar = Au + ((m << 4) + lr) * (LDH / 2) + (ks << 4);
        #pragma unroll
        for (int vv = 0; vv < 8; ++vv) {
          const int t = vv + (hf << 3);
          Af[m].u[vv] = Ar[t];
        }
      }
      #pragma unroll
      for (int m = 0; m < MTILES; ++m)
        acc[m] = __builtin_amdgcn_wmma_f32_16x16x32_f16(
            false, Af[m].v, false, Bf.v, (short)0, acc[m], false, false);
    }
    const float bv = sB[n];                      // LDS, zero-padded: no guard needed
    if (n < Nact) {
      #pragma unroll
      for (int m = 0; m < MTILES; ++m) {
        v8f a = acc[m];
        #pragma unroll
        for (int i = 0; i < 8; ++i) {
          const int row = (m << 4) + (hf << 3) + i;   // C/D layout: M = i + 8*half
          float x = a[i] + bv;
          if (relu) x = fmaxf(x, 0.f);
          Os[row * ldo + n] = x;
        }
      }
    }
  }
}

// ---- head_dim==1 attention: one lane per (b,h); stable softmax over WINDOW ----
__device__ __attribute__((noinline))
void attention(const float* Qs, float* Ts)
{
  for (int t = (int)threadIdx.x; t < BT * NHEAD; t += NTHR) {
    const int b = t / NHEAD, h = t - b * NHEAD;
    float kk[WINDOW], vv[WINDOW];
    #pragma unroll
    for (int j = 0; j < WINDOW; ++j) {
      const float* r = Qs + (j * BT + b) * LDQ;
      kk[j] = r[FEATS + h];
      vv[j] = r[2 * FEATS + h];
    }
    for (int s = 0; s < WINDOW; ++s) {
      const float q = Qs[(s * BT + b) * LDQ + h];
      float mx = -3.4e38f;
      #pragma unroll
      for (int j = 0; j < WINDOW; ++j) mx = fmaxf(mx, q * kk[j]);
      float den = 0.f, num = 0.f;
      #pragma unroll
      for (int j = 0; j < WINDOW; ++j) {
        const float e = __expf(q * kk[j] - mx);
        den += e; num += e * vv[j];
      }
      Ts[(s * BT + b) * LDA + h] = num / den;
    }
  }
}

// ---- A = LayerNorm(A + U) (row-wise over FEATS, eps=1e-5) ----
__device__ __attribute__((noinline))
void add_ln(float* As, const float* Us, int ldu,
            const float* __restrict__ w, const float* __restrict__ b)
{
  UNIF(ldu);
  for (int r = (int)threadIdx.x; r < ROWS; r += NTHR) {
    float* a = As + r * LDA;
    const float* u = Us + r * ldu;
    float tmp[FEATS];
    float mean = 0.f;
    #pragma unroll
    for (int f = 0; f < FEATS; ++f) { tmp[f] = a[f] + u[f]; mean += tmp[f]; }
    mean *= (1.f / FEATS);
    float var = 0.f;
    #pragma unroll
    for (int f = 0; f < FEATS; ++f) { const float d = tmp[f] - mean; var += d * d; }
    var *= (1.f / FEATS);
    const float inv = rsqrtf(var + 1e-5f);
    #pragma unroll
    for (int f = 0; f < FEATS; ++f) a[f] = (tmp[f] - mean) * inv * w[f] + b[f];
  }
}

__device__ __attribute__((noinline))
void store_out(const float* As, float* __restrict__ out, int bbase)
{
  UNIF(bbase);
  for (int idx = (int)threadIdx.x; idx < ROWS * FEATS; idx += NTHR) {
    const int r = idx / FEATS, c = idx - r * FEATS;
    const int s = r / BT, b = r - s * BT;
    const float x = As[r * LDA + c];
    out[(s * BATCHN + bbase + b) * FEATS + c] = 1.f / (1.f + __expf(-x));
  }
}

__global__ void __launch_bounds__(NTHR)
tranad_fused_kernel(Params p)
{
  extern __shared__ float smem[];
  float* sA = smem;                    // activations x / h / y     (80x40 f32)
  float* sM = sA + ROWS * LDA;         // encoder memory            (80x40 f32)
  float* sT = sM + ROWS * LDA;         // attn-out / FFN hidden     (80x40 f32)
  float* sQ = sT + ROWS * LDA;         // q|k|v and GEMM outputs U  (80x120 f32)
  _Float16* sH = (_Float16*)(sQ + ROWS * LDQ);  // GEMM A staging   (80x64  f16)
  _Float16* sW = sH + ROWS * LDH;               // GEMM B staging   (128x64 f16)
  float* sB = (float*)(sW + WROWS * LDH);       // bias staging     (128 f32)
  const int bbase = (int)blockIdx.x * BT;

  // ================= encoder =================
  load_tile(p.src, sA, bbase, 1);                                   __syncthreads();
  cast_rows(sA, FEATS, sH);
  stage_w(p.enc_in_w, p.enc_in_b, FEATS, 3*FEATS, 128, sW, sB);     __syncthreads();
  gemm80h(sH, sW, sB, FEATS, 3*FEATS, sQ, LDQ, 0);                  __syncthreads();
  attention(sQ, sT);                                                __syncthreads();
  cast_rows(sT, FEATS, sH);
  stage_w(p.enc_out_w, p.enc_out_b, FEATS, FEATS, 48, sW, sB);      __syncthreads();
  gemm80h(sH, sW, sB, FEATS, FEATS, sQ, LDQ, 0);                    __syncthreads();
  add_ln(sA, sQ, LDQ, p.enc_ln1_w, p.enc_ln1_b);                    __syncthreads();
  cast_rows(sA, FEATS, sH);
  stage_w(p.enc_lin1_w, p.enc_lin1_b, FEATS, DFF, 16, sW, sB);      __syncthreads();
  gemm80h(sH, sW, sB, FEATS, DFF, sT, LDA, 1);                      __syncthreads();
  cast_rows(sT, DFF, sH);
  stage_w(p.enc_lin2_w, p.enc_lin2_b, DFF, FEATS, 48, sW, sB);      __syncthreads();
  gemm80h(sH, sW, sB, DFF, FEATS, sQ, LDQ, 0);                      __syncthreads();
  add_ln(sA, sQ, LDQ, p.enc_ln2_w, p.enc_ln2_b);                    __syncthreads();
  for (int idx = (int)threadIdx.x; idx < ROWS * LDA; idx += NTHR) sM[idx] = sA[idx];
  __syncthreads();

  // ================= decoder =================
  load_tile(p.tgt, sA, bbase, 0);                                   __syncthreads();
  cast_rows(sA, FEATS, sH);
  stage_w(p.dec_sa_in_w, p.dec_sa_in_b, FEATS, 3*FEATS, 128, sW, sB); __syncthreads();
  gemm80h(sH, sW, sB, FEATS, 3*FEATS, sQ, LDQ, 0);                  __syncthreads();
  attention(sQ, sT);                                                __syncthreads();
  cast_rows(sT, FEATS, sH);
  stage_w(p.dec_sa_out_w, p.dec_sa_out_b, FEATS, FEATS, 48, sW, sB); __syncthreads();
  gemm80h(sH, sW, sB, FEATS, FEATS, sQ, LDQ, 0);                    __syncthreads();
  add_ln(sA, sQ, LDQ, p.dec_ln1_w, p.dec_ln1_b);                    __syncthreads();

  // cross-attention: q from y (sA), k|v from memory (sM)
  cast_rows(sA, FEATS, sH);
  stage_w(p.dec_ca_in_w, p.dec_ca_in_b, FEATS, FEATS, 48, sW, sB);  __syncthreads();
  gemm80h(sH, sW, sB, FEATS, FEATS, sQ, LDQ, 0);                    __syncthreads();
  cast_rows(sM, FEATS, sH);
  stage_w(p.dec_ca_in_w + FEATS*FEATS, p.dec_ca_in_b + FEATS,
          FEATS, 2*FEATS, 80, sW, sB);                              __syncthreads();
  gemm80h(sH, sW, sB, FEATS, 2*FEATS, sQ + FEATS, LDQ, 0);          __syncthreads();
  attention(sQ, sT);                                                __syncthreads();
  cast_rows(sT, FEATS, sH);
  stage_w(p.dec_ca_out_w, p.dec_ca_out_b, FEATS, FEATS, 48, sW, sB); __syncthreads();
  gemm80h(sH, sW, sB, FEATS, FEATS, sQ, LDQ, 0);                    __syncthreads();
  add_ln(sA, sQ, LDQ, p.dec_ln2_w, p.dec_ln2_b);                    __syncthreads();

  cast_rows(sA, FEATS, sH);
  stage_w(p.dec_lin1_w, p.dec_lin1_b, FEATS, DFF, 16, sW, sB);      __syncthreads();
  gemm80h(sH, sW, sB, FEATS, DFF, sT, LDA, 1);                      __syncthreads();
  cast_rows(sT, DFF, sH);
  stage_w(p.dec_lin2_w, p.dec_lin2_b, DFF, FEATS, 48, sW, sB);      __syncthreads();
  gemm80h(sH, sW, sB, DFF, FEATS, sQ, LDQ, 0);                      __syncthreads();
  add_ln(sA, sQ, LDQ, p.dec_ln3_w, p.dec_ln3_b);                    __syncthreads();
  store_out(sA, p.out, bbase);
}

extern "C" void kernel_launch(void* const* d_in, const int* in_sizes, int n_in,
                              void* d_out, int out_size, void* d_ws, size_t ws_size,
                              hipStream_t stream)
{
  Params p;
  p.src        = (const float*)d_in[0];
  p.tgt        = (const float*)d_in[1];
  p.enc_in_w   = (const float*)d_in[2];
  p.enc_in_b   = (const float*)d_in[3];
  p.enc_out_w  = (const float*)d_in[4];
  p.enc_out_b  = (const float*)d_in[5];
  p.enc_lin1_w = (const float*)d_in[6];
  p.enc_lin1_b = (const float*)d_in[7];
  p.enc_lin2_w = (const float*)d_in[8];
  p.enc_lin2_b = (const float*)d_in[9];
  p.enc_ln1_w  = (const float*)d_in[10];
  p.enc_ln1_b  = (const float*)d_in[11];
  p.enc_ln2_w  = (const float*)d_in[12];
  p.enc_ln2_b  = (const float*)d_in[13];
  p.dec_sa_in_w  = (const float*)d_in[14];
  p.dec_sa_in_b  = (const float*)d_in[15];
  p.dec_sa_out_w = (const float*)d_in[16];
  p.dec_sa_out_b = (const float*)d_in[17];
  p.dec_ca_in_w  = (const float*)d_in[18];
  p.dec_ca_in_b  = (const float*)d_in[19];
  p.dec_ca_out_w = (const float*)d_in[20];
  p.dec_ca_out_b = (const float*)d_in[21];
  p.dec_lin1_w = (const float*)d_in[22];
  p.dec_lin1_b = (const float*)d_in[23];
  p.dec_lin2_w = (const float*)d_in[24];
  p.dec_lin2_b = (const float*)d_in[25];
  p.dec_ln1_w  = (const float*)d_in[26];
  p.dec_ln1_b  = (const float*)d_in[27];
  p.dec_ln2_w  = (const float*)d_in[28];
  p.dec_ln2_b  = (const float*)d_in[29];
  p.dec_ln3_w  = (const float*)d_in[30];
  p.dec_ln3_b  = (const float*)d_in[31];
  p.out = (float*)d_out;

  const size_t smem = (size_t)(3 * ROWS * LDA + ROWS * LDQ) * sizeof(float)
                    + (size_t)(ROWS * LDH + WROWS * LDH) * 2u
                    + (size_t)WROWS * sizeof(float);   // 103,936 B (< 320KB/WGP)
  (void)hipFuncSetAttribute((const void*)tranad_fused_kernel,
                            hipFuncAttributeMaxDynamicSharedMemorySize, (int)smem);
  tranad_fused_kernel<<<dim3(BATCHN / BT), dim3(NTHR), smem, stream>>>(p);
}
